// UniMoEAudioSparseMoeBlock_10050223472655
// MI455X (gfx1250) — compile-verified, matrix operand order checked by reading, not run on hardware
//
#include <hip/hip_runtime.h>
#include <hip/hip_bf16.h>
#include <cstdint>
#include <cstddef>

// ---------------- problem constants ----------------
#define BDIM   2
#define SDIM   2048
#define HDIM   1024
#define T_TOK  4096          // B*S
#define EDYN   8
#define NEXP   9             // EDYN + 1 shared
#define IDYN   512
#define IFIX   2048
#define TOPK   4
#define CAPTOK 640           // ceil(T/EDYN * 1.25)
#define MTILE  64            // token rows per FFN block
#define ICHUNK 512           // intermediate chunk (== IDYN; IFIX = 4 chunks)

typedef __bf16 bf16;
typedef __attribute__((ext_vector_type(16))) __bf16 v16bf;
typedef __attribute__((ext_vector_type(8)))  float  v8f;
typedef __attribute__((ext_vector_type(4)))  unsigned int u32x4;
typedef __attribute__((ext_vector_type(8)))  int    i32x8;
typedef __attribute__((ext_vector_type(4)))  int    i32x4;

// ---------------- fp32 -> bf16 conversion ----------------
__global__ void cvt_bf16_kernel(const float* __restrict__ src, bf16* __restrict__ dst, int n) {
    int i = blockIdx.x * blockDim.x + threadIdx.x;
    if (i < n) dst[i] = (bf16)src[i];
}

// ---------------- zero output ----------------
__global__ void zero_kernel(float* __restrict__ out, int n) {
    int i = blockIdx.x * blockDim.x + threadIdx.x;
    if (i < n) out[i] = 0.0f;
}

// ---------------- gating: logits + top-p + sparse mixer + routing weights ----
__global__ void gating_kernel(const float* __restrict__ x,
                              const float* __restrict__ gate_w,
                              float* __restrict__ logits,
                              float* __restrict__ rw) {
    __shared__ float gws[NEXP * HDIM];   // 36 KB
    for (int i = threadIdx.x; i < NEXP * HDIM; i += blockDim.x) gws[i] = gate_w[i];
    __syncthreads();

    int t = blockIdx.x * blockDim.x + threadIdx.x;
    if (t >= T_TOK) return;

    float acc[NEXP];
    #pragma unroll
    for (int e = 0; e < NEXP; ++e) acc[e] = 0.0f;
    const float* xr = x + (size_t)t * HDIM;
    for (int h = 0; h < HDIM; ++h) {
        float xv = xr[h];
        #pragma unroll
        for (int e = 0; e < NEXP; ++e) acc[e] += xv * gws[e * HDIM + h];
    }
    #pragma unroll
    for (int e = 0; e < NEXP; ++e) logits[(size_t)t * NEXP + e] = acc[e];

    // top-p dynamic k from softmax(dyn logits)
    float mx = acc[0];
    #pragma unroll
    for (int e = 1; e < EDYN; ++e) mx = fmaxf(mx, acc[e]);
    float p[EDYN]; float den = 0.0f;
    #pragma unroll
    for (int e = 0; e < EDYN; ++e) { p[e] = __expf(acc[e] - mx); den += p[e]; }
    #pragma unroll
    for (int e = 0; e < EDYN; ++e) p[e] /= den;
    float sp[EDYN];
    #pragma unroll
    for (int e = 0; e < EDYN; ++e) sp[e] = p[e];
    // sort descending (8 elements)
    #pragma unroll
    for (int i = 0; i < EDYN - 1; ++i)
        #pragma unroll
        for (int j = 0; j < EDYN - 1 - i; ++j)
            if (sp[j] < sp[j + 1]) { float tmp = sp[j]; sp[j] = sp[j + 1]; sp[j + 1] = tmp; }
    int cnt = 0; float cs = 0.0f;
    #pragma unroll
    for (int i = 0; i < EDYN; ++i) { cs += sp[i]; if (cs < 0.75f) cnt++; }
    int dyn_k = cnt + 1; if (dyn_k > TOPK) dyn_k = TOPK;

    // sparse mixer (4 rounds)
    float sc[EDYN], msk[EDYN];
    #pragma unroll
    for (int e = 0; e < EDYN; ++e) { sc[e] = acc[e]; msk[e] = acc[e]; }
    float mults[TOPK]; int sels[TOPK];
    #pragma unroll
    for (int kk = 0; kk < TOPK; ++kk) {
        float thr = msk[0]; int sel = 0;
        #pragma unroll
        for (int e = 1; e < EDYN; ++e) if (msk[e] > thr) { thr = msk[e]; sel = e; }
        float d2 = 0.0f, num = 0.0f;
        #pragma unroll
        for (int e = 0; e < EDYN; ++e) {
            float fac = fmaxf(fabsf(sc[e]), fabsf(thr));
            bool jm = (thr - sc[e]) > 0.2f * fac;   // (thr-sc)/fac > 2*JITTER_EPS
            float v = jm ? 0.0f : __expf(msk[e] - thr);
            d2 += v;
            if (e == sel) num = v;
        }
        mults[kk] = num / d2;
        sels[kk] = sel;
        msk[sel] = -__builtin_inff();
    }
    float rwl[EDYN];
    #pragma unroll
    for (int e = 0; e < EDYN; ++e) rwl[e] = 0.0f;
    #pragma unroll
    for (int kk = 0; kk < TOPK; ++kk) if (kk < dyn_k) rwl[sels[kk]] += mults[kk];
    float s = 0.0f;
    #pragma unroll
    for (int e = 0; e < EDYN; ++e) s += rwl[e];
    float inv = 1.0f / s;
    #pragma unroll
    for (int e = 0; e < EDYN; ++e) rw[(size_t)t * EDYN + e] = rwl[e] * inv;
}

// -------- capacity: token-ordered cumsum per expert (wave scan, wave32) -----
__global__ void capacity_kernel(const float* __restrict__ rw,
                                unsigned char* __restrict__ mask) {
    const int e = threadIdx.x >> 5;      // 8 waves -> 8 experts
    const int lane = threadIdx.x & 31;
    int base = 0;
    for (int c = 0; c < T_TOK / 32; ++c) {
        int t = c * 32 + lane;
        bool m = rw[(size_t)t * EDYN + e] > 0.0f;
        unsigned bal = (unsigned)__ballot(m);
        int pre = __popc(bal & ((1u << lane) - 1u));
        bool keep = m && (base + pre + 1 <= CAPTOK);
        mask[(size_t)t * EDYN + e] = keep ? 1 : 0;
        base += __popc(bal);
    }
}

// ---------------- final per-token global weights ----------------
__global__ void weights_kernel(const float* __restrict__ logits,
                               const float* __restrict__ rw,
                               const unsigned char* __restrict__ mask,
                               float* __restrict__ gwt) {
    int t = blockIdx.x * blockDim.x + threadIdx.x;
    if (t >= T_TOK) return;
    float l[NEXP]; int mk[NEXP];
    #pragma unroll
    for (int e = 0; e < NEXP; ++e) l[e] = logits[(size_t)t * NEXP + e];
    #pragma unroll
    for (int e = 0; e < EDYN; ++e) mk[e] = mask[(size_t)t * EDYN + e];
    mk[EDYN] = 1;
    float mx = -__builtin_inff();
    #pragma unroll
    for (int e = 0; e < NEXP; ++e) if (mk[e] && l[e] > mx) mx = l[e];
    float g[NEXP]; float den = 0.0f;
    #pragma unroll
    for (int e = 0; e < NEXP; ++e) { g[e] = mk[e] ? __expf(l[e] - mx) : 0.0f; den += g[e]; }
    float inv = 1.0f / den;
    float sd = 0.0f;
    #pragma unroll
    for (int e = 0; e < EDYN; ++e) sd += g[e] * inv;
    #pragma unroll
    for (int e = 0; e < EDYN; ++e)
        gwt[(size_t)t * NEXP + e] = mk[e] ? rw[(size_t)t * EDYN + e] * sd : 0.0f;
    gwt[(size_t)t * NEXP + EDYN] = g[EDYN] * inv;
}

// ---------------- WMMA expert FFN ----------------
// grid: (T/MTILE, NEXP); block: 256 threads (8 waves); dyn LDS: 192 KB
__global__ __launch_bounds__(256) void moe_ffn_kernel(
    const bf16* __restrict__ xb,
    const bf16* __restrict__ wgb, const bf16* __restrict__ wub, const bf16* __restrict__ wdb,
    const bf16* __restrict__ shg, const bf16* __restrict__ shu, const bf16* __restrict__ shd,
    const float* __restrict__ gwt, float* __restrict__ out) {

    extern __shared__ char smem[];
    bf16* xs = (bf16*)smem;                                // MTILE x HDIM  (128 KB)
    bf16* hs = (bf16*)(smem + (size_t)MTILE * HDIM * 2);   // MTILE x ICHUNK (64 KB)

    const int e    = blockIdx.y;
    const int t0   = blockIdx.x * MTILE;
    const int tid  = threadIdx.x;
    const int wave = tid >> 5;
    const int lane = tid & 31;

    // early-exit for dynamic experts with no routed tokens in this tile
    __shared__ int s_any;
    if (tid == 0) s_any = 0;
    __syncthreads();
    if (e < EDYN) {
        if (tid < MTILE) {
            if (gwt[(size_t)(t0 + tid) * NEXP + e] != 0.0f) atomicOr(&s_any, 1);
        }
        __syncthreads();
        if (s_any == 0) return;
    }

    const bf16 *Wg, *Wu, *Wd;
    int ldI, nIc, wcol;
    if (e < EDYN) {
        Wg = wgb + (size_t)e * IDYN * HDIM;
        Wu = wub + (size_t)e * IDYN * HDIM;
        Wd = wdb + (size_t)e * HDIM * IDYN;
        ldI = IDYN; nIc = 1; wcol = e;
    } else {
        Wg = shg; Wu = shu; Wd = shd;
        ldI = IFIX; nIc = IFIX / ICHUNK; wcol = EDYN;
    }

    // ---- load X tile (MTILE x HDIM bf16) into LDS via Tensor Data Mover ----
#if __has_builtin(__builtin_amdgcn_tensor_load_to_lds)
    if (wave == 0) {
        unsigned long long ga = (unsigned long long)(uintptr_t)(xb + (size_t)t0 * HDIM);
        unsigned int ldsOff = (unsigned int)(uintptr_t)xs;
        u32x4 g0;
        g0[0] = 1u;                                        // count = 1 (valid descriptor)
        g0[1] = ldsOff;                                    // lds_addr
        g0[2] = (unsigned int)(ga & 0xFFFFFFFFull);        // global_addr[31:0]
        g0[3] = (unsigned int)((ga >> 32) & 0x1FFFFFFull) | 0x80000000u; // addr[56:32] | type=2
        i32x8 g1;
        g1[0] = 0x00010000;                                // data_size=1 (2B), wg_mask=0
        g1[1] = (int)((HDIM & 0xFFFF) << 16);              // tensor_dim0 lo16
        g1[2] = (int)((HDIM >> 16) | ((T_TOK & 0xFFFF) << 16)); // dim0 hi | dim1 lo
        g1[3] = (int)((T_TOK >> 16) | (HDIM << 16));       // dim1 hi | tile_dim0 = HDIM
        g1[4] = MTILE;                                     // tile_dim1 = 64, tile_dim2 = 0
        g1[5] = HDIM;                                      // tensor_dim0_stride lo32
        g1[6] = 0;
        g1[7] = 0;
        i32x4 gz; gz[0] = 0; gz[1] = 0; gz[2] = 0; gz[3] = 0;
        i32x8 gz8;
        gz8[0] = 0; gz8[1] = 0; gz8[2] = 0; gz8[3] = 0;
        gz8[4] = 0; gz8[5] = 0; gz8[6] = 0; gz8[7] = 0;
        __builtin_amdgcn_tensor_load_to_lds(g0, g1, gz, gz, gz8, 0);
        __builtin_amdgcn_s_wait_tensorcnt(0);
    }
#else
    {
        const float4* src = (const float4*)(xb + (size_t)t0 * HDIM);
        float4* dst = (float4*)xs;
        for (int i = tid; i < MTILE * HDIM * 2 / 16; i += 256) dst[i] = src[i];
    }
#endif
    __syncthreads();

    const int m    = wave & 3;        // 16-row strip within tile
    const int half = wave >> 2;       // splits N dimension between wave pairs
    const int ksel = (lane >> 4) * 16;
    const int mrow = (lane >> 4) * 8;
    const v8f vzero = {0.f, 0.f, 0.f, 0.f, 0.f, 0.f, 0.f, 0.f};

    for (int ic = 0; ic < nIc; ++ic) {
        const int ibase = ic * ICHUNK;

        // ---------- phase 1: G = X*Wg^T, U = X*Wu^T, h = silu(G)*U -> LDS ----
        const bf16* arow1 = xs + (size_t)(m * 16 + (lane & 15)) * HDIM;
        for (int ncc = 0; ncc < 4; ++ncc) {
            const int n0 = half * 256 + ncc * 64;
            v8f ag[4], au[4];
            #pragma unroll
            for (int j = 0; j < 4; ++j) { ag[j] = vzero; au[j] = vzero; }
            for (int kc = 0; kc < HDIM; kc += 32) {
                v16bf a = *(const v16bf*)(arow1 + kc + ksel);
                #pragma unroll
                for (int j = 0; j < 4; ++j) {
                    const int col = ibase + n0 + j * 16 + (lane & 15);
                    v16bf bg = *(const v16bf*)(Wg + (size_t)col * HDIM + kc + ksel);
                    v16bf bu = *(const v16bf*)(Wu + (size_t)col * HDIM + kc + ksel);
                    ag[j] = __builtin_amdgcn_wmma_f32_16x16x32_bf16(
                        false, a, false, bg, (short)0, ag[j], false, false);
                    au[j] = __builtin_amdgcn_wmma_f32_16x16x32_bf16(
                        false, a, false, bu, (short)0, au[j], false, false);
                }
            }
            #pragma unroll
            for (int j = 0; j < 4; ++j) {
                #pragma unroll
                for (int r = 0; r < 8; ++r) {
                    float g = ag[j][r];
                    float u = au[j][r];
                    float hv = g * (1.0f / (1.0f + __expf(-g))) * u;
                    int M = m * 16 + mrow + r;
                    int N = n0 + j * 16 + (lane & 15);
                    hs[(size_t)M * ICHUNK + N] = (bf16)hv;
                }
            }
        }
        __syncthreads();

        // ---------- phase 2: Y = h * Wd^T, scale by routing weight, atomic add
        const bf16* arow2 = hs + (size_t)(m * 16 + (lane & 15)) * ICHUNK;
        for (int ncc = 0; ncc < 8; ++ncc) {
            const int n0 = half * 512 + ncc * 64;
            v8f ac[4];
            #pragma unroll
            for (int j = 0; j < 4; ++j) ac[j] = vzero;
            for (int kc = 0; kc < ICHUNK; kc += 32) {
                v16bf a = *(const v16bf*)(arow2 + kc + ksel);
                #pragma unroll
                for (int j = 0; j < 4; ++j) {
                    const int col = n0 + j * 16 + (lane & 15);
                    v16bf b = *(const v16bf*)(Wd + (size_t)col * ldI + ibase + kc + ksel);
                    ac[j] = __builtin_amdgcn_wmma_f32_16x16x32_bf16(
                        false, a, false, b, (short)0, ac[j], false, false);
                }
            }
            #pragma unroll
            for (int j = 0; j < 4; ++j) {
                #pragma unroll
                for (int r = 0; r < 8; ++r) {
                    int M = m * 16 + mrow + r;
                    int t = t0 + M;
                    float w = gwt[(size_t)t * NEXP + wcol];
                    if (w != 0.0f) {
                        atomicAdd(&out[(size_t)t * HDIM + n0 + j * 16 + (lane & 15)],
                                  ac[j][r] * w);
                    }
                }
            }
        }
        __syncthreads();
    }
}

// ---------------- launcher ----------------
extern "C" void kernel_launch(void* const* d_in, const int* in_sizes, int n_in,
                              void* d_out, int out_size, void* d_ws, size_t ws_size,
                              hipStream_t stream) {
    (void)in_sizes; (void)n_in; (void)out_size; (void)ws_size;

    const float* x      = (const float*)d_in[0];
    const float* gate_w = (const float*)d_in[1];
    const float* dgw    = (const float*)d_in[2];
    const float* duw    = (const float*)d_in[3];
    const float* ddw    = (const float*)d_in[4];
    const float* sgw    = (const float*)d_in[5];
    const float* suw    = (const float*)d_in[6];
    const float* sdw    = (const float*)d_in[7];
    float* out = (float*)d_out;

    char* ws = (char*)d_ws;
    size_t off = 0;
    auto alloc = [&](size_t bytes) { char* p = ws + off; off = (off + bytes + 255) & ~(size_t)255; return p; };

    bf16* xb   = (bf16*)alloc((size_t)T_TOK * HDIM * 2);
    bf16* wgb  = (bf16*)alloc((size_t)EDYN * IDYN * HDIM * 2);
    bf16* wub  = (bf16*)alloc((size_t)EDYN * IDYN * HDIM * 2);
    bf16* wdb  = (bf16*)alloc((size_t)EDYN * HDIM * IDYN * 2);
    bf16* shgb = (bf16*)alloc((size_t)IFIX * HDIM * 2);
    bf16* shub = (bf16*)alloc((size_t)IFIX * HDIM * 2);
    bf16* shdb = (bf16*)alloc((size_t)HDIM * IFIX * 2);
    float* logits = (float*)alloc((size_t)T_TOK * NEXP * 4);
    float* rw     = (float*)alloc((size_t)T_TOK * EDYN * 4);
    unsigned char* mask = (unsigned char*)alloc((size_t)T_TOK * EDYN);
    float* gwt    = (float*)alloc((size_t)T_TOK * NEXP * 4);

    const int NX  = T_TOK * HDIM;          // 4,194,304
    const int NWE = EDYN * IDYN * HDIM;    // 4,194,304
    const int NSH = IFIX * HDIM;           // 2,097,152

    cvt_bf16_kernel<<<(NX  + 255) / 256, 256, 0, stream>>>(x,   xb,   NX);
    cvt_bf16_kernel<<<(NWE + 255) / 256, 256, 0, stream>>>(dgw, wgb,  NWE);
    cvt_bf16_kernel<<<(NWE + 255) / 256, 256, 0, stream>>>(duw, wub,  NWE);
    cvt_bf16_kernel<<<(NWE + 255) / 256, 256, 0, stream>>>(ddw, wdb,  NWE);
    cvt_bf16_kernel<<<(NSH + 255) / 256, 256, 0, stream>>>(sgw, shgb, NSH);
    cvt_bf16_kernel<<<(NSH + 255) / 256, 256, 0, stream>>>(suw, shub, NSH);
    cvt_bf16_kernel<<<(NSH + 255) / 256, 256, 0, stream>>>(sdw, shdb, NSH);

    gating_kernel<<<T_TOK / 128, 128, 0, stream>>>(x, gate_w, logits, rw);
    capacity_kernel<<<1, 256, 0, stream>>>(rw, mask);
    weights_kernel<<<T_TOK / 256, 256, 0, stream>>>(logits, rw, mask, gwt);
    zero_kernel<<<(NX + 255) / 256, 256, 0, stream>>>(out, NX);

    dim3 grid(T_TOK / MTILE, NEXP);
    size_t ldsBytes = (size_t)MTILE * HDIM * 2 + (size_t)MTILE * ICHUNK * 2; // 192 KB
    moe_ffn_kernel<<<grid, 256, ldsBytes, stream>>>(xb, wgb, wub, wdb,
                                                    shgb, shub, shdb, gwt, out);
}
